// SeparableAttention_86552180949763
// MI455X (gfx1250) — compile-verified, matrix operand order checked by reading, not run on hardware
//
#include <hip/hip_runtime.h>
#include <hip/hip_bf16.h>

// ---------------- problem constants (from reference) ----------------
#define BB   16
#define NN   256
#define TT   50
#define CC   384
#define HH   12
#define DD   32
#define WIN  49
#define KV   (WIN*DD)      // 1568
#define SCALE 0.17677669529663687f   // 1/sqrt(32)

typedef __attribute__((ext_vector_type(16))) __bf16 v16bf;
typedef __attribute__((ext_vector_type(8)))  float  v8f;
typedef unsigned int u32;

// base K of the pair held in vector slot pair v (0..7) for 16-bit A/B fragments
__device__ __forceinline__ int wmma_kpair(int v, int half) {
  return ((v < 4) ? 2 * v : 16 + 2 * (v - 4)) + 8 * half;
}

// A-style fragment from row-major [M x K]: 8 paired (dword) loads, no repacking
__device__ __forceinline__ v16bf load_a_rowmajor(const __bf16* p, int row, int lda,
                                                 int kbase, int lane) {
  int m = lane & 15, half = lane >> 4;
  const __bf16* r = p + (size_t)(row + m) * lda + kbase;
  union { v16bf bf; u32 u[8]; } f;
#pragma unroll
  for (int v = 0; v < 8; ++v) f.u[v] = *(const u32*)(r + wmma_kpair(v, half));
  return f.bf;
}

// packed fragment: lane's 16 bf16 are contiguous (32B) -> b128-class loads
__device__ __forceinline__ v16bf load_packed(const __bf16* p, int lane) {
  return *(const v16bf*)(p + (size_t)lane * 16);
}

__device__ __forceinline__ v8f wmma_bf16(v16bf a, v16bf b, v8f c) {
  return __builtin_amdgcn_wmma_f32_16x16x32_bf16(false, a, false, b, (short)0, c,
                                                 false, false);
}

// ---------------- K0: pack fp32 row-major W[K x N] into B-fragment blocks ----------
// dst block id = ntile*(K/32)+kc ; entry = (block*32 + lane)*16 + i
__global__ void k_packB(const float* __restrict__ Wsrc, __bf16* __restrict__ dst,
                        int K, int N) {
  int idx = blockIdx.x * 256 + threadIdx.x;        // one lane-entry per thread
  int total = (N / 16) * (K / 32) * 32;
  if (idx >= total) return;
  int lane = idx & 31;
  int blk = idx >> 5;
  int kcs = K / 32;
  int kc = blk % kcs, ntile = blk / kcs;
  int nn = lane & 15, half = lane >> 4;
  __bf16* o = dst + (size_t)idx * 16;
#pragma unroll
  for (int i = 0; i < 16; ++i) {
    int k = kc * 32 + wmma_kpair(i >> 1, half) + (i & 1);
    o[i] = (__bf16)Wsrc[(size_t)k * N + ntile * 16 + nn];
  }
}

// ---------------- K1: fused QKV GEMM + window attention per (b,n) ----------------
__global__ __launch_bounds__(256) void k_fused_qkv_attn(
    const float* __restrict__ x, const __bf16* __restrict__ Wqkv,
    float* __restrict__ win_tok, __bf16* __restrict__ pv) {
  __shared__ alignas(32) __bf16 sx[64 * CC];        // 48 KB, rows >= TT zeroed
  __shared__ alignas(32) __bf16 sq[64 * DD];        // 4 KB
  __shared__ alignas(32) __bf16 sk[64 * DD];        // 4 KB
  __shared__ alignas(32) __bf16 svp[2 * 2 * 32 * 16]; // v in packed-B layout, 4 KB
  __shared__ alignas(32) float  sS[64 * 64];        // 16 KB
  __shared__ alignas(32) __bf16 sP[64 * 64];        // 8 KB

  const int bn = blockIdx.x;
  const int b = bn >> 8, n = bn & 255;
  const int tid = threadIdx.x, lane = tid & 31, wv = tid >> 5;
  const float* xt = x + (size_t)bn * TT * CC;

  // stage x tile as bf16 (paired b32 stores)
  for (int i2 = tid; i2 < 64 * CC / 2; i2 += 256) {
    int i = i2 * 2;
    int r = i / CC, c = i - r * CC;
    u32 pkd = 0;
    if (r < TT) {
      union { __bf16 h[2]; u32 u; } uu;
      uu.h[0] = (__bf16)xt[(size_t)r * CC + c];
      uu.h[1] = (__bf16)xt[(size_t)r * CC + c + 1];
      pkd = uu.u;
    }
    *(u32*)(sx + i) = pkd;
  }
  __syncthreads();

  for (int hd = 0; hd < HH; ++hd) {
    // ---- q/k/v for head hd: 3 outs x 4 Mtiles x 2 Ntiles = 24 tiles, K=384 ----
    for (int t = wv; t < 24; t += 8) {
      int nt = t & 1, mt = (t >> 1) & 3, which = t >> 3;
      int ntile = which * 24 + hd * 2 + nt;          // tile column in (3C=1152)/16
      const __bf16* wb = Wqkv + (size_t)ntile * 12 * 32 * 16;
      __builtin_prefetch(wb, 0, 3);
      v8f acc = {};
#pragma unroll
      for (int kc = 0; kc < 12; ++kc) {
        v16bf a  = load_a_rowmajor(sx, mt * 16, CC, kc * 32, lane);
        v16bf bf = load_packed(wb + (size_t)kc * 512, lane);
        acc = wmma_bf16(a, bf, acc);
      }
      int col = nt * 16 + (lane & 15);
      int rbase = mt * 16 + 8 * (lane >> 4);
      if (which == 2) {
        // store v directly in packed-B layout for the O = P @ v GEMM
#pragma unroll
        for (int r = 0; r < 8; ++r) {
          int s = rbase + r;
          int kc32 = s >> 5, kk = s & 31, halfc = (kk >> 3) & 1;
          int km = kk - 8 * halfc;
          int i = (km < 8) ? km : km - 8;
          svp[(((size_t)(nt * 2 + kc32) * 32) + (lane & 15) + 16 * halfc) * 16 + i] =
              (__bf16)acc[r];
        }
      } else {
        __bf16* dst = (which == 0) ? sq : sk;
#pragma unroll
        for (int r = 0; r < 8; ++r) dst[(rbase + r) * DD + col] = (__bf16)acc[r];
      }
    }
    __syncthreads();

    // ---- S = q @ k^T * scale (16 tiles, K=32) ----
    for (int t = wv; t < 16; t += 8) {
      int st = t & 3, mt = t >> 2;
      v16bf a  = load_a_rowmajor(sq, mt * 16, DD, 0, lane);
      v16bf bf = load_a_rowmajor(sk, st * 16, DD, 0, lane);   // k^T via A-style gather
      v8f acc = {};
      acc = wmma_bf16(a, bf, acc);
      int col = st * 16 + (lane & 15);
      int rbase = mt * 16 + 8 * (lane >> 4);
#pragma unroll
      for (int r = 0; r < 8; ++r) sS[(rbase + r) * 64 + col] = acc[r] * SCALE;
    }
    __syncthreads();

    // ---- softmax rows (valid cols [0,TT)) ----
    if (tid < 64) {
      int row = tid;
      if (row < TT) {
        float mx = -1e30f;
        for (int s = 0; s < TT; ++s) mx = fmaxf(mx, sS[row * 64 + s]);
        float sum = 0.f;
        for (int s = 0; s < TT; ++s) sum += __expf(sS[row * 64 + s] - mx);
        float inv = 1.f / sum;
        for (int s = 0; s < TT; ++s)
          sP[row * 64 + s] = (__bf16)(__expf(sS[row * 64 + s] - mx) * inv);
        for (int s = TT; s < 64; ++s) sP[row * 64 + s] = (__bf16)0.0f;
      } else {
        for (int s = 0; s < 64; ++s) sP[row * 64 + s] = (__bf16)0.0f;
      }
    }
    __syncthreads();

    // ---- O = P @ v : 8 tiles, one per wave; K=64 ----
    {
      int mt = wv >> 1, nt = wv & 1;
      v8f acc = {};
#pragma unroll
      for (int kc = 0; kc < 2; ++kc) {
        v16bf a  = load_a_rowmajor(sP, mt * 16, 64, kc * 32, lane);
        v16bf bf = load_packed(svp + (size_t)(nt * 2 + kc) * 32 * 16, lane);
        acc = wmma_bf16(a, bf, acc);
      }
      int j = nt * 16 + (lane & 15);
      int rbase = mt * 16 + 8 * (lane >> 4);
      int c = hd * DD + j;
#pragma unroll
      for (int r = 0; r < 8; ++r) {
        int trow = rbase + r;
        if (trow < TT) {
          float val = acc[r] + xt[(size_t)trow * CC + c];    // +x residual
          if (trow == 0)
            win_tok[(size_t)bn * CC + c] = val;
          else
            pv[((size_t)(b * HH + hd) * NN + n) * KV + (size_t)(trow - 1) * DD + j] =
                (__bf16)val;
        }
      }
    }
    __syncthreads();
  }
}

// ---------------- K2: layernorm + exact gelu on win_tok -> bf16 ----------------
__global__ __launch_bounds__(128) void k_ln_gelu(const float* __restrict__ wt,
                                                 const float* __restrict__ g,
                                                 const float* __restrict__ bta,
                                                 __bf16* __restrict__ out) {
  __shared__ float rs[128], rs2[128];
  int row = blockIdx.x, tid = threadIdx.x;
  const float* p = wt + (size_t)row * CC;
  float s = 0.f, s2 = 0.f;
  for (int c = tid; c < CC; c += 128) { float v = p[c]; s += v; s2 += v * v; }
  rs[tid] = s; rs2[tid] = s2; __syncthreads();
  for (int off = 64; off > 0; off >>= 1) {
    if (tid < off) { rs[tid] += rs[tid + off]; rs2[tid] += rs2[tid + off]; }
    __syncthreads();
  }
  float mean = rs[0] / CC, var = rs2[0] / CC - mean * mean;
  float inv = rsqrtf(var + 1e-5f);
  for (int c = tid; c < CC; c += 128) {
    float y = (p[c] - mean) * inv * g[c] + bta[c];
    float ge = 0.5f * y * (1.f + erff(y * 0.70710678118f));
    out[(size_t)row * CC + c] = (__bf16)ge;
  }
}

// ---------------- K3: qk projection GEMM (4096x384 @ 384x768) ----------------
__global__ __launch_bounds__(256) void k_qk_proj(const __bf16* __restrict__ wt,
                                                 const __bf16* __restrict__ Wqk,
                                                 __bf16* __restrict__ pq,
                                                 __bf16* __restrict__ pk) {
  int mt = blockIdx.x;                       // 0..255
  int lane = threadIdx.x & 31, wv = threadIdx.x >> 5;
  int ntile = blockIdx.y * 8 + wv;           // 0..47
  const __bf16* wb = Wqk + (size_t)ntile * 12 * 32 * 16;
  __builtin_prefetch(wb, 0, 3);
  v8f acc = {};
#pragma unroll
  for (int kc = 0; kc < 12; ++kc) {
    v16bf a = load_a_rowmajor(wt, mt * 16, CC, kc * 32, lane);
    v16bf b = load_packed(wb + (size_t)kc * 512, lane);
    acc = wmma_bf16(a, b, acc);
  }
  int colg = ntile * 16 + (lane & 15);
  int s = colg / CC, rem = colg - s * CC, head = rem >> 5, j = rem & 31;
  __bf16* dst = s ? pk : pq;
  int rbase = mt * 16 + 8 * (lane >> 4);
#pragma unroll
  for (int r = 0; r < 8; ++r) {
    int m = rbase + r, b_ = m >> 8, n_ = m & 255;
    dst[((size_t)(b_ * HH + head) * NN + n_) * DD + j] = (__bf16)acc[r];
  }
}

// ---------------- K4a: pooled scores S = pq @ pk^T * scale ----------------
__global__ __launch_bounds__(256) void k_pool_scores(const __bf16* __restrict__ pq,
                                                     const __bf16* __restrict__ pk,
                                                     float* __restrict__ paf) {
  int bh = blockIdx.x;
  int lane = threadIdx.x & 31, wv = threadIdx.x >> 5;
  const __bf16* q = pq + (size_t)bh * NN * DD;
  const __bf16* k = pk + (size_t)bh * NN * DD;
  for (int t = wv; t < 256; t += 8) {
    int st = t & 15, mt = t >> 4;
    v16bf a = load_a_rowmajor(q, mt * 16, DD, 0, lane);
    v16bf b = load_a_rowmajor(k, st * 16, DD, 0, lane);   // pk^T
    v8f acc = {};
    acc = wmma_bf16(a, b, acc);
    int col = st * 16 + (lane & 15);
    int rbase = mt * 16 + 8 * (lane >> 4);
#pragma unroll
    for (int r = 0; r < 8; ++r)
      paf[(size_t)bh * 65536 + (size_t)(rbase + r) * 256 + col] = acc[r] * SCALE;
  }
}

// ---------------- K4b: row softmax -> bf16 probs ----------------
__global__ __launch_bounds__(256) void k_pool_softmax(const float* __restrict__ paf,
                                                      __bf16* __restrict__ pa) {
  size_t row = (size_t)blockIdx.x * 256 + threadIdx.x;   // 49152 rows
  const float* p = paf + row * 256;
  float mx = -1e30f;
  for (int j = 0; j < 256; ++j) mx = fmaxf(mx, p[j]);
  float sum = 0.f;
  for (int j = 0; j < 256; ++j) sum += __expf(p[j] - mx);
  float inv = 1.f / sum;
  __bf16* o = pa + row * 256;
  for (int j = 0; j < 256; ++j) o[j] = (__bf16)(__expf(p[j] - mx) * inv);
}

// ---------------- K5: po = pa @ pv, + residual, window re-scatter ----------------
__global__ __launch_bounds__(256) void k_pool_apply(const __bf16* __restrict__ pa,
                                                    const __bf16* __restrict__ pv,
                                                    __bf16* __restrict__ rr) {
  __shared__ alignas(32) __bf16 sBp[14 * 8 * 32 * 16];   // packed pv fragments, 112 KB
  int bh = blockIdx.x, b = bh / HH, hd = bh % HH;
  int lane = threadIdx.x & 31, wv = threadIdx.x >> 5;
  const __bf16* A = pa + (size_t)bh * 65536;
  const __bf16* V = pv + (size_t)bh * NN * KV;
  int ncbase = blockIdx.y * 14;              // 98 N-tiles split over gridDim.y==7

  // cooperative pack of B fragments: entries = ntl(14) x kc(8) x lane(32)
  for (int e = threadIdx.x; e < 14 * 8 * 32; e += 256) {
    int ln = e & 31;
    int rest = e >> 5;
    int kc = rest & 7, ntl = rest >> 3;
    int nn = ln & 15, half = ln >> 4;
    const __bf16* src = V + (size_t)(kc * 32) * KV + (size_t)(ncbase + ntl) * 16 + nn;
    __builtin_prefetch(src, 0, 1);
    __bf16* o = sBp + (size_t)e * 16;
#pragma unroll
    for (int i = 0; i < 16; ++i)
      o[i] = src[(size_t)(wmma_kpair(i >> 1, half) + (i & 1)) * KV];
  }
  __syncthreads();

  for (int mi = 0; mi < 2; ++mi) {
    int mt = wv + mi * 8;
    v16bf afr[8];
#pragma unroll
    for (int kc = 0; kc < 8; ++kc)
      afr[kc] = load_a_rowmajor(A, mt * 16, 256, kc * 32, lane);
    for (int ntl = 0; ntl < 14; ++ntl) {
      v8f acc = {};
#pragma unroll
      for (int kc = 0; kc < 8; ++kc) {
        v16bf bf = load_packed(sBp + (size_t)(ntl * 8 + kc) * 32 * 16, lane);
        acc = wmma_bf16(afr[kc], bf, acc);
      }
      int col = (ncbase + ntl) * 16 + (lane & 15);   // 0..1567
      int w = col >> 5, j = col & 31;
      int wh = w / 7, ww = w - wh * 7;
      int rbase = mt * 16 + 8 * (lane >> 4);
#pragma unroll
      for (int r = 0; r < 8; ++r) {
        int n = rbase + r;
        float val = acc[r] + (float)V[(size_t)n * KV + col];   // + attn_x residual
        int gh = n >> 4, gw = n & 15;
        int y = gh * 7 + wh, xx = gw * 7 + ww;
        rr[((size_t)(b * 112 + y) * 112 + xx) * CC + hd * DD + j] = (__bf16)val;
      }
    }
  }
}

// ---------------- K6: final projection (200704x384 @ 384x384) + bias ----------------
__global__ __launch_bounds__(256) void k_final_proj(const __bf16* __restrict__ rr,
                                                    const __bf16* __restrict__ Wp,
                                                    const float* __restrict__ bp,
                                                    float* __restrict__ out) {
  int mt = blockIdx.x;                        // 0..12543
  int lane = threadIdx.x & 31, wv = threadIdx.x >> 5;
  int ntl = blockIdx.y * 8 + wv;              // 0..23
  const __bf16* wb = Wp + (size_t)ntl * 12 * 32 * 16;
  __builtin_prefetch(wb, 0, 3);
  v8f acc = {};
#pragma unroll
  for (int kc = 0; kc < 12; ++kc) {
    v16bf a = load_a_rowmajor(rr, mt * 16, CC, kc * 32, lane);
    v16bf b = load_packed(wb + (size_t)kc * 512, lane);
    acc = wmma_bf16(a, b, acc);
  }
  int col = ntl * 16 + (lane & 15);
  int rbase = mt * 16 + 8 * (lane >> 4);
  float bias = bp[col];
#pragma unroll
  for (int r = 0; r < 8; ++r)
    out[(size_t)(rbase + r) * CC + col] = acc[r] + bias;
}

// ---------------- host-side orchestration ----------------
extern "C" void kernel_launch(void* const* d_in, const int* in_sizes, int n_in,
                              void* d_out, int out_size, void* d_ws, size_t ws_size,
                              hipStream_t stream) {
  (void)in_sizes; (void)n_in; (void)out_size; (void)ws_size;
  const float* x     = (const float*)d_in[0];
  const float* Wqkv  = (const float*)d_in[3];
  const float* ln_g  = (const float*)d_in[4];
  const float* ln_b  = (const float*)d_in[5];
  const float* Wqk   = (const float*)d_in[6];
  const float* Wproj = (const float*)d_in[7];
  const float* bproj = (const float*)d_in[8];
  float* out = (float*)d_out;

  char* w = (char*)d_ws;
  auto alloc = [&](size_t bytes) {
    void* p = (void*)w;
    w += (bytes + 255) & ~(size_t)255;
    return p;
  };
  __bf16* Wqkv_b  = (__bf16*)alloc((size_t)CC * 3 * CC * 2);   // packed fragments
  __bf16* Wqk_b   = (__bf16*)alloc((size_t)CC * 2 * CC * 2);
  __bf16* Wproj_b = (__bf16*)alloc((size_t)CC * CC * 2);
  float*  win_tok = (float*) alloc((size_t)BB * NN * CC * 4);
  __bf16* wtg     = (__bf16*)alloc((size_t)BB * NN * CC * 2);
  __bf16* pq      = (__bf16*)alloc((size_t)BB * HH * NN * DD * 2);
  __bf16* pk      = (__bf16*)alloc((size_t)BB * HH * NN * DD * 2);
  __bf16* pa      = (__bf16*)alloc((size_t)BB * HH * NN * NN * 2);
  __bf16* pv      = (__bf16*)alloc((size_t)BB * HH * NN * KV * 2);
  __bf16* rr      = (__bf16*)alloc((size_t)BB * 112 * 112 * CC * 2);
  float*  paf     = (float*)rr;   // alias: paf fully consumed before rr is written

  k_packB<<<(72 * 12 * 32 + 255) / 256, 256, 0, stream>>>(Wqkv, Wqkv_b, CC, 3 * CC);
  k_packB<<<(48 * 12 * 32 + 255) / 256, 256, 0, stream>>>(Wqk, Wqk_b, CC, 2 * CC);
  k_packB<<<(24 * 12 * 32 + 255) / 256, 256, 0, stream>>>(Wproj, Wproj_b, CC, CC);

  k_fused_qkv_attn<<<BB * NN, 256, 0, stream>>>(x, Wqkv_b, win_tok, pv);
  k_ln_gelu<<<BB * NN, 128, 0, stream>>>(win_tok, ln_g, ln_b, wtg);
  k_qk_proj<<<dim3(256, 6), 256, 0, stream>>>(wtg, Wqk_b, pq, pk);
  k_pool_scores<<<BB * HH, 256, 0, stream>>>(pq, pk, paf);
  k_pool_softmax<<<BB * HH, 256, 0, stream>>>(paf, pa);
  k_pool_apply<<<dim3(BB * HH, 7), 256, 0, stream>>>(pa, pv, rr);
  k_final_proj<<<dim3(12544, 3), 256, 0, stream>>>(rr, Wproj_b, bproj, out);
}
